// LMModel_65068754534439
// MI455X (gfx1250) — matrix operand order repaired and checked by reading, not compiled, not run on hardware
//
#include <hip/hip_runtime.h>

// ---------------------------------------------------------------------------
// LSTM LM for MI455X (gfx1250): L=128, B=64, V=6000, H=1024, G=4H=4096
// ~206 GFLOP total vs ~0.3GB HBM traffic -> compute/latency bound.
// Strategy: bf16 V_WMMA_F32_16X16X32_BF16 everywhere; weights pre-swizzled
// into WMMA A-fragment order (resident in 192MB L2); gates-as-M layout so the
// full LSTM cell update is lane-local (no LDS exchange); persistent kernel
// with an atomic grid barrier for the 128-step serial recurrence.
// ---------------------------------------------------------------------------

#define LL 128
#define BB 64
#define VV 6000
#define HH 1024
#define GG 4096
#define NBLK 16           // persistent blocks in recurrent kernel (= H/64)
#define KT (HH / 32)      // 32 k-tiles of depth 32

typedef __bf16  v16bf __attribute__((ext_vector_type(16)));
typedef float   v8f   __attribute__((ext_vector_type(8)));

__device__ __forceinline__ unsigned short f2bf(float f) {
    unsigned int u = __float_as_uint(f);
    unsigned int r = u + 0x7FFFu + ((u >> 16) & 1u);   // round-to-nearest-even
    return (unsigned short)(r >> 16);
}
__device__ __forceinline__ float bf2f(unsigned short h) {
    return __uint_as_float(((unsigned int)h) << 16);
}
__device__ __forceinline__ float sigf(float x) { return 1.0f / (1.0f + __expf(-x)); }
__device__ __forceinline__ float tanhfast(float x) { return 1.0f - 2.0f / (__expf(2.0f * x) + 1.0f); }
__device__ __forceinline__ v8f v8f_zero() {
    v8f z;
#pragma unroll
    for (int i = 0; i < 8; ++i) z[i] = 0.0f;
    return z;
}
__device__ __forceinline__ v8f wmma_bf16(v16bf a, v16bf b, v8f c) {
    return __builtin_amdgcn_wmma_f32_16x16x32_bf16(false, a, false, b, (short)0, c, false, false);
}

// --------------------------- init: zero scratch ----------------------------
__global__ void zero_kernel(unsigned int* __restrict__ p, int n) {
    int i = blockIdx.x * blockDim.x + threadIdx.x;
    if (i < n) p[i] = 0u;
}

// ------------------ weight fp32 -> bf16 WMMA-A-fragment swizzle ------------
// Per (16g x 32k) tile: 512 bf16 elems, laid out lane-major then element:
//   lane l, elem e : g = gt*16 + (l&15); k = kt*32 + 16*(l>>4) + e
// so each lane's 16 elements are one contiguous 16-element K-run,
// loaded by the GEMM as a single 32B/lane coalesced vector load.
__global__ __launch_bounds__(256) void swizzle_kernel(const float* __restrict__ w,
                                                      unsigned short* __restrict__ out) {
    int idx = blockIdx.x * 256 + threadIdx.x;
    if (idx >= GG * HH) return;
    int e    = idx & 15;
    int l    = (idx >> 4) & 31;
    int tile = idx >> 9;
    int gt = tile / KT, kt = tile % KT;
    int g  = gt * 16 + (l & 15);
    int hi = l >> 4;
    int k  = kt * 32 + 16 * hi + e;
    out[idx] = f2bf(w[g * HH + k]);
}

// --------------- layer-0 input projection: row gather + biases -------------
// x0t[t][g][b] = w_ih0[g][source[t][b]] + b_ih0[g] + b_hh0[g]   (L,G,B) bf16
__global__ __launch_bounds__(256) void embed_kernel(const int* __restrict__ src,
                                                    const float* __restrict__ wih,
                                                    const float* __restrict__ bi,
                                                    const float* __restrict__ bh,
                                                    unsigned short* __restrict__ xout) {
    __shared__ int s[BB];
    int t = blockIdx.x;
    if (threadIdx.x < BB) s[threadIdx.x] = src[t * BB + threadIdx.x];
    __syncthreads();
    for (int idx = threadIdx.x; idx < GG * BB; idx += 256) {
        int g = idx >> 6, b = idx & 63;
        float v = wih[g * VV + s[b]] + bi[g] + bh[g];
        xout[t * GG * BB + idx] = f2bf(v);
    }
}

// ----------------- layer-1 input projection (big bf16 GEMM) ----------------
// x1t[t][g][b] = sum_k w_ih1[g,k] * h1[t*B+b, k]  + b_ih1[g] + b_hh1[g]
// M = G = 4096 (g), N = L*B = 8192 (lb), K = H = 1024.
// Block tile 128x128, 8 waves; wave tile M32 x N64 -> 8 f32 accum tiles.
__global__ __launch_bounds__(256) void proj_kernel(const unsigned short* __restrict__ wswz,
                                                   const unsigned short* __restrict__ h1,
                                                   const float* __restrict__ bi,
                                                   const float* __restrict__ bh,
                                                   unsigned short* __restrict__ xout) {
    const int tid = threadIdx.x;
    const int wave = tid >> 5, lane = tid & 31;
    const int lane16 = lane & 15, hi = lane >> 4;
    const int bx = blockIdx.x & 31;      // 32 tiles along M (G)
    const int by = blockIdx.x >> 5;      // 64 tiles along N (L*B)
    const int wx = wave >> 1;            // 0..3 -> M sub
    const int wy = wave & 1;             // 0..1 -> N sub
    const int m0 = bx * 128 + wx * 32;
    const int n0 = by * 128 + wy * 64;

    v8f acc[2][4];
#pragma unroll
    for (int mt = 0; mt < 2; ++mt)
#pragma unroll
        for (int nt = 0; nt < 4; ++nt) acc[mt][nt] = v8f_zero();

    const int abase = (m0 >> 4) * KT * 512 + lane * 16;   // +mt*KT*512 for mt=1
#pragma unroll 2
    for (int kb = 0; kb < KT; ++kb) {
        v16bf a[2], bfr[4];
#pragma unroll
        for (int mt = 0; mt < 2; ++mt)
            a[mt] = *(const v16bf*)(wswz + abase + mt * KT * 512 + kb * 512);
#pragma unroll
        for (int nt = 0; nt < 4; ++nt) {
            int lb = n0 + nt * 16 + lane16;
            bfr[nt] = *(const v16bf*)(h1 + lb * HH + kb * 32 + 16 * hi);
        }
#pragma unroll
        for (int mt = 0; mt < 2; ++mt)
#pragma unroll
            for (int nt = 0; nt < 4; ++nt)
                acc[mt][nt] = wmma_bf16(a[mt], bfr[nt], acc[mt][nt]);
    }

#pragma unroll
    for (int mt = 0; mt < 2; ++mt) {
#pragma unroll
        for (int r = 0; r < 8; ++r) {
            int g = m0 + mt * 16 + r + 8 * hi;
            float bias = bi[g] + bh[g];
#pragma unroll
            for (int nt = 0; nt < 4; ++nt) {
                int lb = n0 + nt * 16 + lane16;
                int tt = lb >> 6, b = lb & 63;
                xout[tt * GG * BB + g * BB + b] = f2bf(acc[mt][nt][r] + bias);
            }
        }
    }
}

// --------------------- persistent recurrent LSTM layer ---------------------
// Computes, per step t:  Z^T[g,b] = x[t,g,b] + sum_k W[g,k] * h_prev[b,k]
// A = swizzled W (16x32 bf16 frags), B = h_prev rows (16 contiguous K / lane).
// C layout: m = vgpr + 8*(lane>=16), n = lane&15  ->  for fixed (lane, vgpr)
// the 4 gate accumulators hold i,f,g,o of the SAME (j,b): cell update is
// entirely lane-local, no LDS. 16 blocks own 64 hidden cols each; a global
// atomic barrier sequences the 128 steps.
template <bool FINAL>
__global__ __launch_bounds__(256) void lstm_layer_kernel(const unsigned short* __restrict__ wswz,
                                                         const unsigned short* __restrict__ xproj,
                                                         const unsigned short* __restrict__ hzero,
                                                         unsigned short* __restrict__ hseq,
                                                         float* __restrict__ cstate,
                                                         float* __restrict__ outf32,
                                                         unsigned int* __restrict__ counter) {
    const int tid = threadIdx.x;
    const int wave = tid >> 5, lane = tid & 31;
    const int lane16 = lane & 15, hi = lane >> 4;
    const int bid = blockIdx.x;          // 16 blocks, each owns 64 hidden cols
    const int wx = wave >> 1;            // 0..3 -> 16-col sub-tile
    const int wy = wave & 1;             // 0..1 -> batch half
    const int j16 = bid * 64 + wx * 16;  // hidden col base of this wave
    const int b0 = wy * 32;

    int abase[4];
#pragma unroll
    for (int q = 0; q < 4; ++q) {
        int gt = (q * HH + j16) >> 4;
        abase[q] = gt * KT * 512 + lane * 16;
    }

    for (int t = 0; t < LL; ++t) {
        const unsigned short* hsrc =
            (t == 0) ? hzero : (FINAL ? hseq + ((t - 1) & 1) * BB * HH : hseq + (t - 1) * BB * HH);
        unsigned short* hdst = FINAL ? hseq + (t & 1) * BB * HH : hseq + t * BB * HH;

        v8f acc[4][2];
#pragma unroll
        for (int q = 0; q < 4; ++q)
#pragma unroll
            for (int t2 = 0; t2 < 2; ++t2) acc[q][t2] = v8f_zero();

#pragma unroll 2
        for (int kb = 0; kb < KT; ++kb) {
            v16bf a[4], bfr[2];
#pragma unroll
            for (int q = 0; q < 4; ++q)
                a[q] = *(const v16bf*)(wswz + abase[q] + kb * 512);
#pragma unroll
            for (int t2 = 0; t2 < 2; ++t2) {
                int b = b0 + t2 * 16 + lane16;
                bfr[t2] = *(const v16bf*)(hsrc + b * HH + kb * 32 + 16 * hi);
            }
#pragma unroll
            for (int q = 0; q < 4; ++q)
#pragma unroll
                for (int t2 = 0; t2 < 2; ++t2)
                    acc[q][t2] = wmma_bf16(a[q], bfr[t2], acc[q][t2]);
        }

        // lane-local LSTM cell update
#pragma unroll
        for (int t2 = 0; t2 < 2; ++t2) {
            int b = b0 + t2 * 16 + lane16;
#pragma unroll
            for (int r = 0; r < 8; ++r) {
                int j = j16 + r + 8 * hi;
                int xbase = t * GG * BB + j * BB + b;
                float zi = acc[0][t2][r] + bf2f(xproj[xbase + 0 * HH * BB]);
                float zf = acc[1][t2][r] + bf2f(xproj[xbase + 1 * HH * BB]);
                float zg = acc[2][t2][r] + bf2f(xproj[xbase + 2 * HH * BB]);
                float zo = acc[3][t2][r] + bf2f(xproj[xbase + 3 * HH * BB]);
                float cold = cstate[j * BB + b];
                float cn = sigf(zf) * cold + sigf(zi) * tanhfast(zg);
                float hn = sigf(zo) * tanhfast(cn);
                cstate[j * BB + b] = cn;
                hdst[b * HH + j] = f2bf(hn);
                if (FINAL) outf32[(t * BB + b) * HH + j] = hn;
            }
        }

        // grid-wide step barrier (16 resident blocks)
        __threadfence();
        __syncthreads();
        if (tid == 0) {
            __hip_atomic_fetch_add(counter, 1u, __ATOMIC_ACQ_REL, __HIP_MEMORY_SCOPE_AGENT);
            unsigned target = (unsigned)(NBLK * (t + 1));
            while (__hip_atomic_load(counter, __ATOMIC_ACQUIRE, __HIP_MEMORY_SCOPE_AGENT) < target) {
                __builtin_amdgcn_s_sleep(1);
            }
        }
        __syncthreads();
        __threadfence();
    }
}

// ---------------------------------------------------------------------------
extern "C" void kernel_launch(void* const* d_in, const int* in_sizes, int n_in,
                              void* d_out, int out_size, void* d_ws, size_t ws_size,
                              hipStream_t stream) {
    const int*   src   = (const int*)d_in[0];
    const float* w_ih0 = (const float*)d_in[1];
    const float* w_hh0 = (const float*)d_in[2];
    const float* b_ih0 = (const float*)d_in[3];
    const float* b_hh0 = (const float*)d_in[4];
    const float* w_ih1 = (const float*)d_in[5];
    const float* w_hh1 = (const float*)d_in[6];
    const float* b_ih1 = (const float*)d_in[7];
    const float* b_hh1 = (const float*)d_in[8];
    float* out = (float*)d_out;

    char* ws = (char*)d_ws;
    // zeroed region: [counters(256B) | c0 | c1 | hzero]
    size_t off = 0;
    unsigned int*   cnt   = (unsigned int*)(ws + off);      off += 256;
    float*          c0    = (float*)(ws + off);             off += (size_t)HH * BB * 4;
    float*          c1    = (float*)(ws + off);             off += (size_t)HH * BB * 4;
    unsigned short* hzero = (unsigned short*)(ws + off);    off += (size_t)BB * HH * 2;
    const size_t zero_dwords = off / 4;
    // working buffers
    unsigned short* xproj = (unsigned short*)(ws + off);    off += (size_t)LL * GG * BB * 2; // 64MB, reused x0 then x1
    unsigned short* h1seq = (unsigned short*)(ws + off);    off += (size_t)LL * BB * HH * 2; // 16MB
    unsigned short* hbuf  = (unsigned short*)(ws + off);    off += (size_t)2 * BB * HH * 2;
    unsigned short* wswz0 = (unsigned short*)(ws + off);    off += (size_t)GG * HH * 2;
    unsigned short* wswz1 = (unsigned short*)(ws + off);    off += (size_t)GG * HH * 2;
    unsigned short* wswzI = (unsigned short*)(ws + off);    off += (size_t)GG * HH * 2;
    (void)ws_size; (void)in_sizes; (void)n_in; (void)out_size;

    // 1) zero counters / cell states / h0
    zero_kernel<<<(unsigned)((zero_dwords + 255) / 256), 256, 0, stream>>>(cnt, (int)zero_dwords);

    // 2) weight bf16 + WMMA-fragment swizzle (stay resident in L2 afterwards)
    const int swgrid = (GG * HH + 255) / 256;
    swizzle_kernel<<<swgrid, 256, 0, stream>>>(w_hh0, wswz0);
    swizzle_kernel<<<swgrid, 256, 0, stream>>>(w_hh1, wswz1);
    swizzle_kernel<<<swgrid, 256, 0, stream>>>(w_ih1, wswzI);

    // 3) layer-0 input projection = row gather + biases
    embed_kernel<<<LL, 256, 0, stream>>>(src, w_ih0, b_ih0, b_hh0, xproj);

    // 4) layer-0 recurrence (persistent, 16 blocks)
    lstm_layer_kernel<false><<<NBLK, 256, 0, stream>>>(wswz0, xproj, hzero, h1seq, c0, nullptr, cnt + 0);

    // 5) layer-1 input projection: big bf16 WMMA GEMM (4096 x 8192 x 1024)
    proj_kernel<<<32 * 64, 256, 0, stream>>>(wswzI, h1seq, b_ih1, b_hh1, xproj);

    // 6) layer-1 recurrence, fp32 hidden states to d_out
    lstm_layer_kernel<true><<<NBLK, 256, 0, stream>>>(wswz1, xproj, hzero, hbuf, c1, out, cnt + 1);
}
